// Nl_9216999817662
// MI455X (gfx1250) — matrix operand order repaired and checked by reading, not compiled, not run on hardware
//
#include <hip/hip_runtime.h>

typedef unsigned short u16;
typedef __attribute__((ext_vector_type(16))) __bf16 v16bf;
typedef __attribute__((ext_vector_type(8)))  float  v8f;
typedef __attribute__((ext_vector_type(4)))  unsigned int u32x4;
typedef __attribute__((ext_vector_type(8)))  int i32x8;
typedef __attribute__((ext_vector_type(4)))  int i32x4;

#define N_SP  4096
#define D_KQ  128
#define C_IN  256
#define C_OUT 256
#define NB    8

#define KPITCH 136   // 32-key K tile row pitch in u16 (256B data + 16B TDM pad)
#define VPITCH 40    // V tile row pitch in u16 (64B data + 16B TDM pad)

struct FragU { uint4 lo, hi; };

__device__ __forceinline__ u16 f2bf(float f) {
  unsigned u = __builtin_bit_cast(unsigned, f);
  u += 0x7FFFu + ((u >> 16) & 1u);           // round-to-nearest-even
  return (u16)(u >> 16);
}

// A operand (16x32 bf16): lane = row (lane%16); khalf selects K {0..7,16..23} vs {8..15,24..31}
__device__ __forceinline__ v16bf ldA16(const u16* rowbase, int khalf) {
  const u16* p = rowbase + khalf * 8;
  FragU f;
  f.lo = *reinterpret_cast<const uint4*>(p);
  f.hi = *reinterpret_cast<const uint4*>(p + 16);
  return __builtin_bit_cast(v16bf, f);
}

// B operand (32x16 bf16): lane = col (lane%16); khalf selects K {0..7,16..23} vs {8..15,24..31}
__device__ __forceinline__ v16bf ldB16(const u16* rowbase, int khalf) {
  const u16* p = rowbase + khalf * 8;
  FragU f;
  f.lo = *reinterpret_cast<const uint4*>(p);
  f.hi = *reinterpret_cast<const uint4*>(p + 16);
  return __builtin_bit_cast(v16bf, f);
}

__device__ __forceinline__ v8f wmma_bf16(v16bf a, v16bf b, v8f c) {
  return __builtin_amdgcn_wmma_f32_16x16x32_bf16(false, a, false, b, (short)0, c,
                                                 false, false);
}

// ---- DPP 16-lane reductions (never cross the 16-lane halves) ----------------
template <int CTRL>
__device__ __forceinline__ float dpp_xf(float x) {
  int xi = __builtin_bit_cast(int, x);
  int r  = __builtin_amdgcn_update_dpp(xi, xi, CTRL, 0xf, 0xf, false);
  return __builtin_bit_cast(float, r);
}
__device__ __forceinline__ float redmax16(float x) {
  x = fmaxf(x, dpp_xf<0xB1>(x));    // quad_perm [1,0,3,2]  (^1)
  x = fmaxf(x, dpp_xf<0x4E>(x));    // quad_perm [2,3,0,1]  (^2)
  x = fmaxf(x, dpp_xf<0x141>(x));   // row_half_mirror      (^7)
  x = fmaxf(x, dpp_xf<0x140>(x));   // row_mirror           (^15)
  return x;
}
__device__ __forceinline__ float redsum16(float x) {
  x += dpp_xf<0xB1>(x);
  x += dpp_xf<0x4E>(x);
  x += dpp_xf<0x141>(x);
  x += dpp_xf<0x140>(x);
  return x;
}

// ---- Tensor Data Mover: one descriptor load, groups per CDNA5 D# layout -----
__device__ __forceinline__ void tdm_load(unsigned lds_off, unsigned long long ga,
                                         i32x8 g1) {
  u32x4 g0;
  g0[0] = 1u;                                     // count=1 (valid user D#)
  g0[1] = lds_off;                                // LDS byte address
  g0[2] = (unsigned)ga;                           // global_addr[31:0]
  g0[3] = (unsigned)(ga >> 32) | (2u << 30);      // global_addr[56:32] | type=2
  i32x4 z4 = {0, 0, 0, 0};
#if __clang_major__ >= 23
  i32x8 z8 = {0, 0, 0, 0, 0, 0, 0, 0};
  __builtin_amdgcn_tensor_load_to_lds(g0, g1, z4, z4, z8, 0);
#else
  __builtin_amdgcn_tensor_load_to_lds(g0, g1, z4, z4, 0);
#endif
}

// K tile: 1D, 8KB contiguous (32 rows x 256B), data_size=8B, pad 16B per 256B
__device__ __forceinline__ i32x8 g1_ktile() {
  i32x8 g;
  g[0] = (3 << 16) | (1 << 20) | (5 << 22) | (3 << 25); // size=8B,pad on,int=256B,amt=16B
  g[1] = (int)(1024u << 16);   // tensor_dim0[15:0]=1024 (8B units)
  g[2] = (1 << 16);            // tensor_dim0 hi=0 | tensor_dim1=1
  g[3] = (int)(1024u << 16);   // tensor_dim1 hi=0 | tile_dim0=1024 (8KB)
  g[4] = 0;                    // tile_dim1=0 (1D) | tile_dim2=0
  g[5] = 1024;                 // tensor_dim0_stride
  g[6] = 0; g[7] = 0;
  return g;
}
// V tile: 2D, 8 x 128 (64B rows, 8KB global pitch), pad 16B per 64B row
__device__ __forceinline__ i32x8 g1_vtile() {
  i32x8 g;
  g[0] = (3 << 16) | (1 << 20) | (3 << 22) | (3 << 25); // size=8B,pad on,int=64B,amt=16B
  g[1] = (int)(1024u << 16);   // tensor_dim0 = 1024 units (full 4096-key row)
  g[2] = (128 << 16);          // tensor_dim1 = 128 feature rows
  g[3] = (8 << 16);            // tile_dim0 = 8 units (32 keys)
  g[4] = 128;                  // tile_dim1 = 128
  g[5] = 1024;                 // tensor_dim0_stride = 1024 units (8192B)
  g[6] = 0; g[7] = 0;
  return g;
}

// ---------------------------------------------------------------------------
// Kernel 1: kq/v projections (fp32 compute, bf16 outputs in WMMA-friendly layouts)
// ---------------------------------------------------------------------------
__global__ __launch_bounds__(256) void proj_kernel(
    const float* __restrict__ x, const float* __restrict__ w_kq,
    const float* __restrict__ b_kq, const float* __restrict__ w_v,
    const float* __restrict__ b_v, u16* __restrict__ kqb, u16* __restrict__ vb) {
  __shared__ float xs[C_IN * 32];    // x tile [256 c][32 n]  (32 KB)
  __shared__ float wsh[32 * C_IN];   // weight chunk [32 k][256 c] (32 KB)

  const int tid = threadIdx.x;
  const int b   = blockIdx.x >> 7;
  const int n0  = (blockIdx.x & 127) * 32;

  const float4* xsrc = reinterpret_cast<const float4*>(x);
  float4* xs4 = reinterpret_cast<float4*>(xs);
#pragma unroll
  for (int i = 0; i < 8; ++i) {
    int f = tid + 256 * i;
    int c = f >> 3, nq = f & 7;
    xs4[f] = xsrc[(size_t)(b * C_IN + c) * (N_SP / 4) + (n0 >> 2) + nq];
  }
  __syncthreads();

  const int n = tid & 31;
  const int g = tid >> 5;
  float4* wsh4 = reinterpret_cast<float4*>(wsh);

  for (int chunk = 0; chunk < 8; ++chunk) {
    const float* wsrc = (chunk < 4) ? (w_kq + (size_t)chunk * 32 * C_IN)
                                    : (w_v + (size_t)(chunk - 4) * 32 * C_IN);
    const float* bias = (chunk < 4) ? (b_kq + chunk * 32) : (b_v + (chunk - 4) * 32);
    const float4* wsrc4 = reinterpret_cast<const float4*>(wsrc);
#pragma unroll
    for (int i = 0; i < 8; ++i) wsh4[tid + 256 * i] = wsrc4[tid + 256 * i];
    __syncthreads();

    float acc[4];
#pragma unroll
    for (int j = 0; j < 4; ++j) acc[j] = bias[g * 4 + j];
#pragma unroll 4
    for (int c = 0; c < C_IN; ++c) {
      float xv = xs[c * 32 + n];
#pragma unroll
      for (int j = 0; j < 4; ++j)
        acc[j] += wsh[(g * 4 + j) * C_IN + c] * xv;
    }
#pragma unroll
    for (int j = 0; j < 4; ++j) {
      int krow = g * 4 + j;
      if (chunk < 4) {
        int k = chunk * 32 + krow;
        kqb[(size_t)(b * N_SP + n0 + n) * D_KQ + k] = f2bf(acc[j]);
      } else {
        int k = (chunk - 4) * 32 + krow;
        vb[(size_t)(b * D_KQ + k) * N_SP + n0 + n] = f2bf(acc[j]);
      }
    }
    __syncthreads();
  }
}

// ---------------------------------------------------------------------------
// Kernel 2: fused flash attention. K/V tiles staged to LDS by the TDM
// (double-buffered, TENSORcnt synchronized), P transposed via per-wave LDS,
// softmax row reductions in DPP.
// ---------------------------------------------------------------------------
__global__ __launch_bounds__(256) void attn_kernel(
    const u16* __restrict__ kqb, const u16* __restrict__ vb,
    u16* __restrict__ ctxb) {
  __shared__ __align__(16) u16 ksh[2][32 * KPITCH];   // 2 x 8704 B
  __shared__ __align__(16) u16 vsh[2][128 * VPITCH];  // 2 x 10240 B
  __shared__ __align__(16) u16 pbuf[8][16 * 56];      // per-wave P tile (112B rows)

  const int lane  = threadIdx.x & 31;
  const int wave  = threadIdx.x >> 5;
  const int b     = blockIdx.x >> 5;
  const int q0    = (blockIdx.x & 31) * 128 + wave * 16;
  const int khalf = lane >> 4;
  const int ln    = lane & 15;

  const u16* kq_b = kqb + (size_t)b * N_SP * D_KQ;
  const u16* v_b  = vb + (size_t)b * D_KQ * N_SP;
  u16* pb = pbuf[wave];

  const unsigned ksh_off0 = (unsigned)(unsigned long long)(uintptr_t)&ksh[0][0];
  const unsigned ksh_off1 = (unsigned)(unsigned long long)(uintptr_t)&ksh[1][0];
  const unsigned vsh_off0 = (unsigned)(unsigned long long)(uintptr_t)&vsh[0][0];
  const unsigned vsh_off1 = (unsigned)(unsigned long long)(uintptr_t)&vsh[1][0];
  const i32x8 gk = g1_ktile();
  const i32x8 gv = g1_vtile();

  // resident Q fragments (A operand, 4 x K32 over d=128)
  v16bf qf[4];
  {
    const u16* qrow = kq_b + (size_t)(q0 + ln) * D_KQ;
#pragma unroll
    for (int i = 0; i < 4; ++i) qf[i] = ldA16(qrow + i * 32, khalf);
  }

  v8f acc[8];
#pragma unroll
  for (int g = 0; g < 8; ++g)
#pragma unroll
    for (int r = 0; r < 8; ++r) acc[g][r] = 0.0f;

  float mrow[8], lrow[8];
#pragma unroll
  for (int r = 0; r < 8; ++r) { mrow[r] = -3.0e38f; lrow[r] = 0.0f; }

  const float LOG2E = 1.4426950408889634f;
  const int NT = N_SP / 32;                  // 128 key tiles

  if (wave == 0) {                           // preload tile 0
    tdm_load(ksh_off0, (unsigned long long)(uintptr_t)kq_b, gk);
    tdm_load(vsh_off0, (unsigned long long)(uintptr_t)v_b, gv);
  }

#pragma unroll 1
  for (int it = 0; it < NT; ++it) {
    const int cur = it & 1;
    if (wave == 0) {
      if (it + 1 < NT) {                     // prefetch next tile into other buffer
        int kt1 = (it + 1) * 32;
        tdm_load(cur ? ksh_off0 : ksh_off1,
                 (unsigned long long)(uintptr_t)(kq_b + (size_t)kt1 * D_KQ), gk);
        tdm_load(cur ? vsh_off0 : vsh_off1,
                 (unsigned long long)(uintptr_t)(v_b + kt1), gv);
        __builtin_amdgcn_s_wait_tensorcnt(2);  // tile `it` complete
      } else {
        __builtin_amdgcn_s_wait_tensorcnt(0);
      }
    }
    __syncthreads();                         // publish tile `it`

    const u16* kt_sh = ksh[cur];
    const u16* vt_sh = vsh[cur];

    // S[16 x 32] = Q @ K^T (K from LDS)
    v8f S0, S1;
#pragma unroll
    for (int r = 0; r < 8; ++r) { S0[r] = 0.0f; S1[r] = 0.0f; }
#pragma unroll
    for (int i = 0; i < 4; ++i) {
      S0 = wmma_bf16(qf[i], ldB16(kt_sh + (size_t)ln * KPITCH + i * 32, khalf), S0);
      S1 = wmma_bf16(qf[i], ldB16(kt_sh + (size_t)(16 + ln) * KPITCH + i * 32, khalf), S1);
    }

    // online softmax: DPP row reductions (VALU only)
    float p0[8], p1[8];
#pragma unroll
    for (int r = 0; r < 8; ++r) {
      float s  = redmax16(fmaxf(S0[r], S1[r]));
      float mn = fmaxf(mrow[r], s);
      float rescale = exp2f((mrow[r] - mn) * LOG2E);
      mrow[r] = mn;
      float e0 = exp2f((S0[r] - mn) * LOG2E);
      float e1 = exp2f((S1[r] - mn) * LOG2E);
      p0[r] = e0; p1[r] = e1;
      lrow[r] = lrow[r] * rescale + redsum16(e0 + e1);
#pragma unroll
      for (int g = 0; g < 8; ++g) acc[g][r] *= rescale;
    }

    // C-layout -> A-layout through per-wave LDS tile
#pragma unroll
    for (int r = 0; r < 8; ++r) {
      int row = r + khalf * 8;
      pb[row * 56 + ln]      = f2bf(p0[r]);
      pb[row * 56 + 16 + ln] = f2bf(p1[r]);
    }
    v16bf pf = ldA16(pb + ln * 56, khalf);

    // ctx += P[16x32] @ V[32x128] (V from LDS)
#pragma unroll
    for (int g = 0; g < 8; ++g)
      acc[g] = wmma_bf16(pf, ldB16(vt_sh + (size_t)(g * 16 + ln) * VPITCH, khalf),
                         acc[g]);

    __syncthreads();                         // all waves done with buf `cur`
  }

  // normalize and emit ctx (bf16, row-major [B,N,128])
  u16* c_b = ctxb + (size_t)b * N_SP * D_KQ;
#pragma unroll
  for (int r = 0; r < 8; ++r) {
    float inv = 1.0f / lrow[r];
    int row = q0 + r + khalf * 8;
#pragma unroll
    for (int g = 0; g < 8; ++g)
      c_b[(size_t)row * D_KQ + g * 16 + ln] = f2bf(acc[g][r] * inv);
  }
}

// ---------------------------------------------------------------------------
// Kernel 3: out = w_o @ ctx^T + b_o (WMMA, w_o -> bf16 A-frags on the fly)
// ---------------------------------------------------------------------------
__global__ __launch_bounds__(256) void outproj_kernel(
    const u16* __restrict__ ctxb, const float* __restrict__ w_o,
    const float* __restrict__ b_o, float* __restrict__ out) {
  const int lane  = threadIdx.x & 31;
  const int wave  = threadIdx.x >> 5;
  const int b     = blockIdx.x >> 8;
  const int n0    = (blockIdx.x & 255) * 16;
  const int khalf = lane >> 4;
  const int ln    = lane & 15;
  const u16* ctx_b = ctxb + (size_t)b * N_SP * D_KQ;

#pragma unroll
  for (int t = 0; t < 2; ++t) {
    int o0 = (wave * 2 + t) * 16;
    v8f acc;
#pragma unroll
    for (int r = 0; r < 8; ++r) acc[r] = 0.0f;
#pragma unroll
    for (int i = 0; i < 4; ++i) {
      const float* wp = w_o + (size_t)(o0 + ln) * D_KQ + i * 32 + khalf * 8;
      v16bf a;
#pragma unroll
      for (int e = 0; e < 8; ++e) {
        a[e]     = __builtin_bit_cast(__bf16, f2bf(wp[e]));
        a[8 + e] = __builtin_bit_cast(__bf16, f2bf(wp[16 + e]));
      }
      v16bf bb = ldB16(ctx_b + (size_t)(n0 + ln) * D_KQ + i * 32, khalf);
      acc = wmma_bf16(a, bb, acc);
    }
#pragma unroll
    for (int r = 0; r < 8; ++r) {
      int o = o0 + r + khalf * 8;
      out[((size_t)b * C_OUT + o) * N_SP + n0 + ln] = acc[r] + b_o[o];
    }
  }
}

// ---------------------------------------------------------------------------
extern "C" void kernel_launch(void* const* d_in, const int* in_sizes, int n_in,
                              void* d_out, int out_size, void* d_ws, size_t ws_size,
                              hipStream_t stream) {
  (void)in_sizes; (void)n_in; (void)out_size; (void)ws_size;
  const float* x    = (const float*)d_in[0];
  const float* w_kq = (const float*)d_in[1];
  const float* b_kq = (const float*)d_in[2];
  const float* w_v  = (const float*)d_in[3];
  const float* b_v  = (const float*)d_in[4];
  const float* w_o  = (const float*)d_in[5];
  const float* b_o  = (const float*)d_in[6];
  float* out = (float*)d_out;

  u16* kqb  = (u16*)d_ws;                                   // 8 MB
  u16* vb   = kqb + (size_t)NB * N_SP * D_KQ;               // 8 MB
  u16* ctxb = vb + (size_t)NB * D_KQ * N_SP;                // 8 MB

  proj_kernel<<<NB * 128, 256, 0, stream>>>(x, w_kq, b_kq, w_v, b_v, kqb, vb);
  attn_kernel<<<NB * 32, 256, 0, stream>>>(kqb, vb, ctxb);
  outproj_kernel<<<NB * 256, 256, 0, stream>>>(ctxb, w_o, b_o, out);
}
